// VanillaRNN_2628519985319
// MI455X (gfx1250) — matrix operand order repaired and checked by reading, not compiled, not run on hardware
//
#include <hip/hip_runtime.h>
#include <math.h>

typedef __attribute__((ext_vector_type(2)))  float   v2f;
typedef __attribute__((ext_vector_type(8)))  float   v8f;
typedef __attribute__((ext_vector_type(16))) __bf16  v16bf;

#define VOCAB 32000
#define EMB   1024
#define HID   1024
#define BATCH 8
#define SEQ   512
#define MROWS (BATCH * SEQ)   // 4096
#define SCAN_BLOCKS 8

// ---------------------------------------------------------------------------
// bf16 helpers (round-to-nearest-even split of fp32 into hi+lo bf16)
// ---------------------------------------------------------------------------
__device__ __forceinline__ unsigned short f2bf(float f) {
    unsigned u = __float_as_uint(f);
    u = u + 0x7FFFu + ((u >> 16) & 1u);
    return (unsigned short)(u >> 16);
}
__device__ __forceinline__ float bf2f(unsigned short h) {
    return __uint_as_float(((unsigned)h) << 16);
}
// split 8 consecutive floats into packed-bf16 hi and lo uint4's
__device__ __forceinline__ void split8(const float* f, uint4& hv, uint4& lv) {
    unsigned h[4], l[4];
#pragma unroll
    for (int i = 0; i < 4; ++i) {
        unsigned short h0 = f2bf(f[2 * i]), h1 = f2bf(f[2 * i + 1]);
        float l0 = f[2 * i] - bf2f(h0), l1 = f[2 * i + 1] - bf2f(h1);
        h[i] = (unsigned)h0 | ((unsigned)h1 << 16);
        l[i] = (unsigned)f2bf(l0) | ((unsigned)f2bf(l1) << 16);
    }
    hv = make_uint4(h[0], h[1], h[2], h[3]);
    lv = make_uint4(l[0], l[1], l[2], l[3]);
}

union FragU {
    v16bf v;
    uint4 q[2];
};

__device__ __forceinline__ v8f wmma_bf16(v16bf a, v16bf b, v8f c) {
    return __builtin_amdgcn_wmma_f32_16x16x32_bf16(false, a, false, b, (short)0, c,
                                                   false, false);
}
__device__ __forceinline__ v8f wmma_f32(v2f a, v2f b, v8f c) {
    return __builtin_amdgcn_wmma_f32_16x16x4_f32(false, a, false, b, (short)0, c,
                                                 false, false);
}

// ---------------------------------------------------------------------------
// Split-precision bf16x3 GEMM:  C[M,N] = A[M,K] * B[N,K]^T (+ bias[N])
//   fp32 inputs, fp32 accumulation; each fp32 split into hi/lo bf16 at staging.
//   D += Ah*Bh + Ah*Bl + Al*Bh   (lo*lo dropped; ~2^-16 relative error)
// Workgroup tile 128(M) x 64(N), 256 threads = 8 waves, wave tile 32x32,
// K-slab 32 -> 12 v_wmma_f32_16x16x32_bf16 per wave per slab.
// GATHER: A row m comes from Abase + rowids[m]*K (fused embedding lookup).
// ---------------------------------------------------------------------------
template <bool GATHER>
__global__ __launch_bounds__(256) void gemm_bf16x3(
    const float* __restrict__ Abase, const int* __restrict__ rowids,
    const float* __restrict__ Bmat, const float* __restrict__ bias,
    float* __restrict__ C, int M, int N, int K) {
    // pitch 40 bf16 (80 B): rows 16B-aligned, 16-lane reads hit all 64 banks once
    __shared__ unsigned short AtH[128][40];
    __shared__ unsigned short AtL[128][40];
    __shared__ unsigned short BtH[64][40];
    __shared__ unsigned short BtL[64][40];

    const int tid    = threadIdx.x;
    const int lane   = tid & 31;
    const int lane16 = lane & 15;
    const int hi     = lane >> 4;
    const int wid    = tid >> 5;   // 0..7
    const int waveM  = wid >> 1;   // 0..3
    const int waveN  = wid & 1;    // 0..1

    const int N0 = blockIdx.x * 64;
    const int M0 = blockIdx.y * 128;

    v8f acc00 = {0,0,0,0,0,0,0,0};
    v8f acc01 = {0,0,0,0,0,0,0,0};
    v8f acc10 = {0,0,0,0,0,0,0,0};
    v8f acc11 = {0,0,0,0,0,0,0,0};

    // A staging map: 2 threads/row, 16 floats each
    const int a_mi = tid >> 1;
    const int a_kh = (tid & 1) * 16;
    long arow;
    if (GATHER) arow = (long)rowids[M0 + a_mi] * (long)K;
    else        arow = (long)(M0 + a_mi) * (long)K;
    // B staging map: 4 threads/row, 8 floats each (row-major copy, no transpose)
    const int b_n  = tid >> 2;
    const int b_k8 = (tid & 3) * 8;
    const long brow = (long)(N0 + b_n) * (long)K;

    const int ra0 = waveM * 32;
    const int ra1 = waveM * 32 + 16;
    const int cb0 = waveN * 32;
    const int cb1 = waveN * 32 + 16;

    float ra[16], rb[8];
    // prologue: fetch slab 0
    {
        float4 p0 = *(const float4*)(Abase + arow + a_kh);
        float4 p1 = *(const float4*)(Abase + arow + a_kh + 4);
        float4 p2 = *(const float4*)(Abase + arow + a_kh + 8);
        float4 p3 = *(const float4*)(Abase + arow + a_kh + 12);
        ra[0]=p0.x; ra[1]=p0.y; ra[2]=p0.z; ra[3]=p0.w;
        ra[4]=p1.x; ra[5]=p1.y; ra[6]=p1.z; ra[7]=p1.w;
        ra[8]=p2.x; ra[9]=p2.y; ra[10]=p2.z; ra[11]=p2.w;
        ra[12]=p3.x; ra[13]=p3.y; ra[14]=p3.z; ra[15]=p3.w;
        float4 q0 = *(const float4*)(Bmat + brow + b_k8);
        float4 q1 = *(const float4*)(Bmat + brow + b_k8 + 4);
        rb[0]=q0.x; rb[1]=q0.y; rb[2]=q0.z; rb[3]=q0.w;
        rb[4]=q1.x; rb[5]=q1.y; rb[6]=q1.z; rb[7]=q1.w;
    }

    for (int ko = 0; ko < K; ko += 32) {
        // stage current slab into LDS with hi/lo split
        uint4 hv, lv;
        split8(&ra[0], hv, lv);
        *(uint4*)&AtH[a_mi][a_kh]     = hv;  *(uint4*)&AtL[a_mi][a_kh]     = lv;
        split8(&ra[8], hv, lv);
        *(uint4*)&AtH[a_mi][a_kh + 8] = hv;  *(uint4*)&AtL[a_mi][a_kh + 8] = lv;
        split8(&rb[0], hv, lv);
        *(uint4*)&BtH[b_n][b_k8]      = hv;  *(uint4*)&BtL[b_n][b_k8]      = lv;
        __syncthreads();

        // software pipeline: fetch next slab while WMMAs consume LDS;
        // prefetch (global_prefetch_b8) two slabs ahead to cover HBM latency
        if (ko + 32 < K) {
            const float* ap = Abase + arow + (ko + 32);
            float4 p0 = *(const float4*)(ap + a_kh);
            float4 p1 = *(const float4*)(ap + a_kh + 4);
            float4 p2 = *(const float4*)(ap + a_kh + 8);
            float4 p3 = *(const float4*)(ap + a_kh + 12);
            ra[0]=p0.x; ra[1]=p0.y; ra[2]=p0.z; ra[3]=p0.w;
            ra[4]=p1.x; ra[5]=p1.y; ra[6]=p1.z; ra[7]=p1.w;
            ra[8]=p2.x; ra[9]=p2.y; ra[10]=p2.z; ra[11]=p2.w;
            ra[12]=p3.x; ra[13]=p3.y; ra[14]=p3.z; ra[15]=p3.w;
            const float* bp = Bmat + brow + (ko + 32);
            float4 q0 = *(const float4*)(bp + b_k8);
            float4 q1 = *(const float4*)(bp + b_k8 + 4);
            rb[0]=q0.x; rb[1]=q0.y; rb[2]=q0.z; rb[3]=q0.w;
            rb[4]=q1.x; rb[5]=q1.y; rb[6]=q1.z; rb[7]=q1.w;
            if (ko + 64 < K) {
                __builtin_prefetch(Abase + arow + ko + 64 + a_kh, 0, 0);
                __builtin_prefetch(Bmat + brow + ko + 64 + b_k8, 0, 0);
            }
        }

        // load fragments (layouts per ISA 7.12.2, all 16B-aligned b128 LDS reads)
        FragU aH0, aH1, aL0, aL1, bH0, bH1, bL0, bL1;
        aH0.q[0] = *(const uint4*)&AtH[ra0 + lane16][hi * 8];
        aH0.q[1] = *(const uint4*)&AtH[ra0 + lane16][16 + hi * 8];
        aL0.q[0] = *(const uint4*)&AtL[ra0 + lane16][hi * 8];
        aL0.q[1] = *(const uint4*)&AtL[ra0 + lane16][16 + hi * 8];
        aH1.q[0] = *(const uint4*)&AtH[ra1 + lane16][hi * 8];
        aH1.q[1] = *(const uint4*)&AtH[ra1 + lane16][16 + hi * 8];
        aL1.q[0] = *(const uint4*)&AtL[ra1 + lane16][hi * 8];
        aL1.q[1] = *(const uint4*)&AtL[ra1 + lane16][16 + hi * 8];
        bH0.q[0] = *(const uint4*)&BtH[cb0 + lane16][hi * 8];
        bH0.q[1] = *(const uint4*)&BtH[cb0 + lane16][16 + hi * 8];
        bL0.q[0] = *(const uint4*)&BtL[cb0 + lane16][hi * 8];
        bL0.q[1] = *(const uint4*)&BtL[cb0 + lane16][16 + hi * 8];
        bH1.q[0] = *(const uint4*)&BtH[cb1 + lane16][hi * 8];
        bH1.q[1] = *(const uint4*)&BtH[cb1 + lane16][16 + hi * 8];
        bL1.q[0] = *(const uint4*)&BtL[cb1 + lane16][hi * 8];
        bL1.q[1] = *(const uint4*)&BtL[cb1 + lane16][16 + hi * 8];

        // 12 bf16 WMMAs: hi*hi + hi*lo + lo*hi for each of 4 subtiles
        acc00 = wmma_bf16(aH0.v, bH0.v, acc00);
        acc00 = wmma_bf16(aH0.v, bL0.v, acc00);
        acc00 = wmma_bf16(aL0.v, bH0.v, acc00);
        acc01 = wmma_bf16(aH0.v, bH1.v, acc01);
        acc01 = wmma_bf16(aH0.v, bL1.v, acc01);
        acc01 = wmma_bf16(aL0.v, bH1.v, acc01);
        acc10 = wmma_bf16(aH1.v, bH0.v, acc10);
        acc10 = wmma_bf16(aH1.v, bL0.v, acc10);
        acc10 = wmma_bf16(aL1.v, bH0.v, acc10);
        acc11 = wmma_bf16(aH1.v, bH1.v, acc11);
        acc11 = wmma_bf16(aH1.v, bL1.v, acc11);
        acc11 = wmma_bf16(aL1.v, bH1.v, acc11);
        __syncthreads();
    }

    // epilogue (C/D layout: VGPR v -> M=v (+8 for hi lanes), N=lane16)
#pragma unroll
    for (int i = 0; i < 2; ++i) {
        v8f* accRow0 = (i == 0) ? &acc00 : &acc10;
        v8f* accRow1 = (i == 0) ? &acc01 : &acc11;
#pragma unroll
        for (int v = 0; v < 8; ++v) {
            const int row = M0 + waveM * 32 + i * 16 + v + hi * 8;
            const int c0  = N0 + waveN * 32 + lane16;
            const int c1  = c0 + 16;
            float r0 = (*accRow0)[v];
            float r1 = (*accRow1)[v];
            if (bias) { r0 += bias[c0]; r1 += bias[c1]; }
            C[(long)row * (long)N + c0] = r0;
            C[(long)row * (long)N + c1] = r1;
        }
    }
}

// ---------------------------------------------------------------------------
// Persistent RNN scan: all 512 timesteps in ONE launch.
//   h_{t+1} = tanh(xW[:,t,:] + h_t @ W_hh^T + b_hh)
// 8 blocks x 8 waves = 64 waves; wave gw owns hidden columns [gw*16, gw*16+16).
// Exact fp32 WMMA (error-sensitive 512-step recurrence). Inter-step sync is a
// device-wide monotonic atomic barrier (8 co-resident blocks) instead of 512
// kernel launches. h ping-pong buffers padded to 16 rows (rows 8..15 dummy).
// ---------------------------------------------------------------------------
__global__ __launch_bounds__(256) void rnn_scan(
    const float* __restrict__ xW, const float* __restrict__ W_hh,
    const float* __restrict__ b_hh, float* __restrict__ hbuf /* [2][16][HID] */,
    float* __restrict__ hs, unsigned* __restrict__ barrier_ctr) {
    const int tid    = threadIdx.x;
    const int lane   = tid & 31;
    const int lane16 = lane & 15;
    const int hi     = lane >> 4;
    const int gw     = blockIdx.x * 8 + (tid >> 5);  // 0..63
    const int N0     = gw * 16;
    const int Nc     = N0 + lane16;

    const float* bp  = W_hh + (long)Nc * HID;        // loop-invariant B row
    const float  bb  = b_hh[Nc];

    for (int t = 0; t < SEQ; ++t) {
        const float* hp = hbuf + ((t & 1) ? 16 * HID : 0);
        float*       hc = hbuf + ((t & 1) ? 0 : 16 * HID);
        const float* ap = hp + lane16 * HID;

        v8f acc = {0,0,0,0,0,0,0,0};
#pragma unroll 4
        for (int k0 = 0; k0 < HID; k0 += 4) {
            const int ks = k0 + hi * 2;
            v2f a = *(const v2f*)(ap + ks);
            v2f b = *(const v2f*)(bp + ks);
            acc = wmma_f32(a, b, acc);
        }

#pragma unroll
        for (int v = 0; v < 8; ++v) {
            const int Mr = v + hi * 8;               // >=8 is padding
            float xv = 0.0f;
            if (Mr < BATCH) xv = xW[((long)(Mr * SEQ + t)) * HID + Nc];
            float r = tanhf(acc[v] + xv + bb);
            hc[Mr * HID + Nc] = r;
            if (Mr < BATCH) hs[((long)(Mr * SEQ + t)) * HID + Nc] = r;
        }

        // ---- device-wide step barrier (monotonic counter, no reset race) ----
        __syncthreads();                              // block done writing h
        if (tid == 0) {
            __threadfence();                          // release h stores (DEV)
            __hip_atomic_fetch_add(barrier_ctr, 1u, __ATOMIC_RELEASE,
                                   __HIP_MEMORY_SCOPE_AGENT);
            const unsigned target = (unsigned)SCAN_BLOCKS * (unsigned)(t + 1);
            while (__hip_atomic_load(barrier_ctr, __ATOMIC_ACQUIRE,
                                     __HIP_MEMORY_SCOPE_AGENT) < target) {
                __builtin_amdgcn_s_sleep(1);
            }
        }
        __syncthreads();
        __threadfence();                              // acquire: see fresh h
    }
}

__global__ void zero_f32(float* p, int n) {
    int i = blockIdx.x * 256 + threadIdx.x;
    if (i < n) p[i] = 0.0f;
}

__global__ void copy_hidden(const float* __restrict__ h, float* __restrict__ out) {
    int i = blockIdx.x * 256 + threadIdx.x;   // 8192 = rows 0..7 of padded h
    out[i] = h[i];
}

extern "C" void kernel_launch(void* const* d_in, const int* in_sizes, int n_in,
                              void* d_out, int out_size, void* d_ws, size_t ws_size,
                              hipStream_t stream) {
    const int*   x    = (const int*)d_in[0];
    const float* emb  = (const float*)d_in[1];
    const float* W_xh = (const float*)d_in[2];
    const float* W_hh = (const float*)d_in[3];
    const float* b_hh = (const float*)d_in[4];
    const float* W_fc = (const float*)d_in[5];
    const float* b_fc = (const float*)d_in[6];
    float* out = (float*)d_out;

    float* xW   = (float*)d_ws;                // [4096,1024]
    float* hs   = xW + (long)MROWS * HID;      // [4096,1024]
    float* hbuf = hs + (long)MROWS * HID;      // [2][16][1024] ping/pong
    unsigned* ctr = (unsigned*)(hbuf + 2 * 16 * HID);  // barrier counter

    // zero both h buffers + barrier counter (contiguous; 0.0f == 0u bit pattern)
    zero_f32<<<(2 * 16 * HID + 1 + 255) / 256, 256, 0, stream>>>(
        hbuf, 2 * 16 * HID + 1);

    // xW = gather(emb, x) @ W_xh^T  (bf16x3 split GEMM, fused gather)
    dim3 gA(HID / 64, MROWS / 128);
    gemm_bf16x3<true><<<gA, 256, 0, stream>>>(emb, x, W_xh, nullptr, xW,
                                              MROWS, HID, EMB);

    // whole 512-step recurrence in one persistent launch
    rnn_scan<<<SCAN_BLOCKS, 256, 0, stream>>>(xW, W_hh, b_hh, hbuf, hs, ctr);

    // out = hs @ W_fc^T + b_fc  (dominant GEMM, bf16x3 split)
    dim3 gC(VOCAB / 64, MROWS / 128);
    gemm_bf16x3<false><<<gC, 256, 0, stream>>>(hs, nullptr, W_fc, b_fc, out,
                                               MROWS, VOCAB, HID);

    // final hidden state: step 511 wrote buffer 0 (t odd -> hc = hbuf[0])
    copy_hidden<<<(BATCH * HID) / 256, 256, 0, stream>>>(hbuf,
                                                         out + (long)MROWS * VOCAB);
}
